// Net_33560874451329
// MI455X (gfx1250) — matrix operand order repaired and checked by reading, not compiled, not run on hardware
//
#include <hip/hip_runtime.h>
#include <hip/hip_fp16.h>
#include <math.h>

#define BB   4
#define LL   1024
#define DM   512
#define ED   1024
#define NS   16
#define DCONV 4
#define DTR  32
#define ROWS (BB*LL)          // 4096 token rows

#define WM 2                  // 16-row tiles per wave  (32 rows)
#define WN 4                  // 16-col tiles per wave  (64 cols)

typedef __attribute__((ext_vector_type(16))) _Float16 v16h;
typedef __attribute__((ext_vector_type(8)))  _Float16 v8h;
typedef __attribute__((ext_vector_type(8)))  float    v8f;

// ---------------- f32 -> f16 conversion ----------------
__global__ void cvt_f32_f16(const float* __restrict__ s, _Float16* __restrict__ d, int n) {
    int i = blockIdx.x * blockDim.x + threadIdx.x;
    if (i < n) d[i] = (_Float16)s[i];
}

// ---------------- RMSNorm -> f16 ----------------
__global__ void rmsnorm_f16(const float* __restrict__ x, const float* __restrict__ w,
                            _Float16* __restrict__ out) {
    __shared__ float red[256];
    int row = blockIdx.x;
    const float* xr = x + (size_t)row * DM;
    float ss = 0.f;
    for (int c = threadIdx.x; c < DM; c += 256) { float v = xr[c]; ss += v * v; }
    red[threadIdx.x] = ss; __syncthreads();
    for (int off = 128; off > 0; off >>= 1) {
        if (threadIdx.x < off) red[threadIdx.x] += red[threadIdx.x + off];
        __syncthreads();
    }
    float scale = rsqrtf(red[0] / (float)DM + 1e-5f);
    for (int c = threadIdx.x; c < DM; c += 256)
        out[(size_t)row * DM + c] = (_Float16)(xr[c] * scale * w[c]);
}

// ---------------- WMMA GEMM: C[M,N] = A[M,K] * Bw[N,K]^T ----------------
// One wave computes a 32x64 macro-tile (2x4 WMMA tiles) stepping K by 32:
// 8 v_wmma_f32_16x16x32_f16 per k-step vs 12 b128 fragment loads (A reused
// across WN, B reused across WM).  Macro-tiles are M-fastest so adjacent
// waves share the weight panel (Bw) in L2.
// mode 0: C = acc   mode 1: C = softplus(acc + bias[col])   mode 2: C += acc
__global__ void wmma_gemm(const _Float16* __restrict__ A, const _Float16* __restrict__ Bw,
                          float* __restrict__ C, int M, int N, int K,
                          const float* __restrict__ bias, int mode) {
    int wave = threadIdx.x >> 5;
    int lane = threadIdx.x & 31;
    int mtiles = M >> 5;                   // macro tiles (32 rows each)
    int ntiles = N >> 6;                   // macro tiles (64 cols each)
    int t = blockIdx.x * (blockDim.x >> 5) + wave;
    if (t >= mtiles * ntiles) return;      // wave-uniform: EXEC stays all-ones
    int tm = (t % mtiles) << 5;
    int tn = (t / mtiles) << 6;
    int half = lane >> 4;                  // lane group 0..15 / 16..31
    int r    = lane & 15;

    // A fragment rows tm+r (+16); k = kstep + half*8 + {0..7, 16..23}
    const _Float16* arow = A  + (size_t)(tm + r) * K + half * 8;
    // B fragment rows tn+r (+16,+32,+48); k = kstep + half*8 + {0..7, 16..23}
    const _Float16* brow = Bw + (size_t)(tn + r) * K + half * 8;
    size_t rstride = (size_t)16 * K;

    v8f acc[WM][WN] = {};
    for (int k = 0; k < K; k += 32) {
        v16h av[WM], bv[WN];
#pragma unroll
        for (int i = 0; i < WM; ++i) {
            const _Float16* p = arow + (size_t)i * rstride + k;
            v8h a0 = *(const v8h*)p;
            v8h a1 = *(const v8h*)(p + 16);
#pragma unroll
            for (int q = 0; q < 8; ++q) { av[i][q] = a0[q]; av[i][q + 8] = a1[q]; }
        }
#pragma unroll
        for (int j = 0; j < WN; ++j) {
            const _Float16* p = brow + (size_t)j * rstride + k;
            v8h b0 = *(const v8h*)p;
            v8h b1 = *(const v8h*)(p + 16);
#pragma unroll
            for (int q = 0; q < 8; ++q) { bv[j][q] = b0[q]; bv[j][q + 8] = b1[q]; }
        }
#pragma unroll
        for (int i = 0; i < WM; ++i)
#pragma unroll
            for (int j = 0; j < WN; ++j)
                acc[i][j] = __builtin_amdgcn_wmma_f32_16x16x32_f16(
                    false, av[i], false, bv[j], (short)0, acc[i][j], false, false);
    }

#pragma unroll
    for (int i = 0; i < WM; ++i) {
#pragma unroll
        for (int j = 0; j < WN; ++j) {
            int col = tn + j * 16 + r;
#pragma unroll
            for (int rr = 0; rr < 8; ++rr) {
                int row = tm + i * 16 + rr + half * 8;   // VGPR rr: M=rr / M=8+rr
                size_t idx = (size_t)row * N + col;
                float v = acc[i][j][rr];
                if (mode == 1) {
                    v += bias[col];
                    v = (v > 20.f) ? v : log1pf(__expf(v));    // softplus
                    C[idx] = v;
                } else if (mode == 2) {
                    C[idx] += v;                               // residual accumulate
                } else {
                    C[idx] = v;
                }
            }
        }
    }
}

// ---------------- causal depthwise conv1d + bias + SiLU ----------------
__global__ void conv_silu(const float* __restrict__ xz, const float* __restrict__ cw,
                          const float* __restrict__ cb, float* __restrict__ xin,
                          _Float16* __restrict__ xin16) {
    int idx = blockIdx.x * blockDim.x + threadIdx.x;     // ROWS*ED
    if (idx >= ROWS * ED) return;
    int e   = idx & (ED - 1);
    int row = idx >> 10;                                  // ED == 1024
    int l   = row & (LL - 1);
    float acc = cb[e];
#pragma unroll
    for (int j = 0; j < DCONV; ++j) {
        int ls = l - (DCONV - 1) + j;
        if (ls >= 0)
            acc += cw[e * DCONV + j] * xz[(size_t)(row - (DCONV - 1) + j) * (2 * ED) + e];
    }
    float s = acc / (1.f + __expf(-acc));                 // SiLU
    xin[idx] = s;
    xin16[idx] = (_Float16)s;
}

// ---------------- extract delta slice (dbc[:, :32]) -> f16 ----------------
__global__ void extract_delta(const float* __restrict__ dbc, _Float16* __restrict__ d16) {
    int i = blockIdx.x * blockDim.x + threadIdx.x;       // ROWS*DTR
    if (i >= ROWS * DTR) return;
    int row = i >> 5, c = i & 31;
    d16[i] = (_Float16)dbc[(size_t)row * 64 + c];
}

// ---------------- selective scan: recurrent over L, 1 thread per (b,e) ----------------
__global__ void scan_kernel(const float* __restrict__ delta, const float* __restrict__ dbc,
                            const float* __restrict__ xin, const float* __restrict__ A_log,
                            const float* __restrict__ Dv, float* __restrict__ ypre) {
    __shared__ float sBC[32];
    int b = blockIdx.x / (ED / 256);
    int e = (blockIdx.x % (ED / 256)) * 256 + threadIdx.x;
    float Ar[NS], h[NS];
#pragma unroll
    for (int n = 0; n < NS; ++n) { Ar[n] = -__expf(A_log[e * NS + n]); h[n] = 0.f; }
    float Dd = Dv[e];
    for (int l = 0; l < LL; ++l) {
        size_t row = (size_t)b * LL + l;
        if (threadIdx.x < 32) sBC[threadIdx.x] = dbc[row * 64 + 32 + threadIdx.x];
        __syncthreads();
        float dt = delta[row * ED + e];
        float xv = xin[row * ED + e];
        float dx = dt * xv;
        float y  = 0.f;
#pragma unroll
        for (int n = 0; n < NS; ++n) {
            h[n] = __expf(dt * Ar[n]) * h[n] + dx * sBC[n];
            y   += h[n] * sBC[16 + n];
        }
        ypre[row * ED + e] = y + Dd * xv;
        __syncthreads();
    }
}

// ---------------- gate: y16 = (ypre * silu(z)) as f16 ----------------
__global__ void gate_kernel(const float* __restrict__ ypre, const float* __restrict__ xz,
                            _Float16* __restrict__ y16) {
    int i = blockIdx.x * blockDim.x + threadIdx.x;       // ROWS*ED
    if (i >= ROWS * ED) return;
    int row = i >> 10, e = i & (ED - 1);
    float z = xz[(size_t)row * (2 * ED) + ED + e];
    float g = z / (1.f + __expf(-z));
    y16[i] = (_Float16)(ypre[i] * g);
}

// ---------------- final FC + sigmoid ----------------
__global__ void fc_sigmoid(const float* __restrict__ x, const float* __restrict__ fw,
                           const float* __restrict__ fb, float* __restrict__ out) {
    __shared__ float red[256];
    int row = blockIdx.x;
    float s = 0.f;
    for (int c = threadIdx.x; c < DM; c += 256) s += x[(size_t)row * DM + c] * fw[c];
    red[threadIdx.x] = s; __syncthreads();
    for (int off = 128; off > 0; off >>= 1) {
        if (threadIdx.x < off) red[threadIdx.x] += red[threadIdx.x + off];
        __syncthreads();
    }
    if (threadIdx.x == 0) out[row] = 1.f / (1.f + __expf(-(red[0] + fb[0])));
}

// =====================================================================
extern "C" void kernel_launch(void* const* d_in, const int* in_sizes, int n_in,
                              void* d_out, int out_size, void* d_ws, size_t ws_size,
                              hipStream_t stream) {
    const float* x_in   = (const float*)d_in[0];
    const float* norm_w = (const float*)d_in[1];
    const float* in_w   = (const float*)d_in[2];   // [2,2048,512]
    const float* conv_w = (const float*)d_in[3];   // [2,1024,1,4]
    const float* conv_b = (const float*)d_in[4];   // [2,1024]
    const float* xp_w   = (const float*)d_in[5];   // [2,64,1024]
    const float* dt_w   = (const float*)d_in[6];   // [2,1024,32]
    const float* dt_b   = (const float*)d_in[7];   // [2,1024]
    const float* A_log  = (const float*)d_in[8];   // [2,1024,16]
    const float* Dp     = (const float*)d_in[9];   // [2,1024]
    const float* out_w  = (const float*)d_in[10];  // [2,512,1024]
    const float* fc_w   = (const float*)d_in[11];  // [1,512]
    const float* fc_b   = (const float*)d_in[12];  // [1]
    float* out = (float*)d_out;

    char* ws = (char*)d_ws;
    auto alloc = [&](size_t bytes) -> char* {
        char* p = ws; ws += (bytes + 255) & ~(size_t)255; return p;
    };
    float*    x_cur   = (float*)   alloc((size_t)ROWS * DM * 4);
    _Float16* xn16    = (_Float16*)alloc((size_t)ROWS * DM * 2);
    float*    xz      = (float*)   alloc((size_t)ROWS * 2 * ED * 4);
    float*    xin     = (float*)   alloc((size_t)ROWS * ED * 4);
    _Float16* xin16   = (_Float16*)alloc((size_t)ROWS * ED * 2);
    float*    dbc     = (float*)   alloc((size_t)ROWS * 64 * 4);
    _Float16* delta16 = (_Float16*)alloc((size_t)ROWS * DTR * 2);
    float*    delta   = (float*)   alloc((size_t)ROWS * ED * 4);
    float*    ypre    = (float*)   alloc((size_t)ROWS * ED * 4);
    _Float16* y16     = (_Float16*)alloc((size_t)ROWS * ED * 2);
    _Float16* win16   = (_Float16*)alloc((size_t)2 * ED * DM * 2);
    _Float16* wxp16   = (_Float16*)alloc((size_t)(DTR + 2 * NS) * ED * 2);
    _Float16* wdt16   = (_Float16*)alloc((size_t)ED * DTR * 2);
    _Float16* wout16  = (_Float16*)alloc((size_t)DM * ED * 2);

    hipMemcpyAsync(x_cur, x_in, (size_t)ROWS * DM * 4, hipMemcpyDeviceToDevice, stream);

    auto gemm = [&](const _Float16* A, const _Float16* Bw, float* C,
                    int M, int N, int K, const float* bias, int mode) {
        int tiles = (M >> 5) * (N >> 6);              // 32x64 macro-tiles
        int blocks = (tiles + 7) / 8;                 // 8 waves (256 thr) per block
        wmma_gemm<<<blocks, 256, 0, stream>>>(A, Bw, C, M, N, K, bias, mode);
    };
    auto cvt = [&](const float* s, _Float16* d, int n) {
        cvt_f32_f16<<<(n + 255) / 256, 256, 0, stream>>>(s, d, n);
    };

    for (int i = 0; i < 2; ++i) {
        // weight conversion to f16 (deterministic each call)
        cvt(in_w  + (size_t)i * 2 * ED * DM, win16,  2 * ED * DM);
        cvt(xp_w  + (size_t)i * 64 * ED,     wxp16,  64 * ED);
        cvt(dt_w  + (size_t)i * ED * DTR,    wdt16,  ED * DTR);
        cvt(out_w + (size_t)i * DM * ED,     wout16, DM * ED);

        // RMSNorm -> f16
        rmsnorm_f16<<<ROWS, 256, 0, stream>>>(x_cur, norm_w + (size_t)i * DM, xn16);

        // in_proj: xz[4096,2048] = xn @ W_in^T
        gemm(xn16, win16, xz, ROWS, 2 * ED, DM, nullptr, 0);

        // causal depthwise conv + SiLU
        conv_silu<<<(ROWS * ED) / 256, 256, 0, stream>>>(
            xz, conv_w + (size_t)i * ED * DCONV, conv_b + (size_t)i * ED, xin, xin16);

        // x_proj: dbc[4096,64] = xin @ W_xp^T
        gemm(xin16, wxp16, dbc, ROWS, 64, ED, nullptr, 0);

        // delta slice -> f16
        extract_delta<<<(ROWS * DTR) / 256, 256, 0, stream>>>(dbc, delta16);

        // dt_proj + softplus(.+dt_b): delta[4096,1024]
        gemm(delta16, wdt16, delta, ROWS, ED, DTR, dt_b + (size_t)i * ED, 1);

        // selective scan (recurrent over L)
        scan_kernel<<<BB * (ED / 256), 256, 0, stream>>>(
            delta, dbc, xin, A_log + (size_t)i * ED * NS, Dp + (size_t)i * ED, ypre);

        // gating
        gate_kernel<<<(ROWS * ED) / 256, 256, 0, stream>>>(ypre, xz, y16);

        // out_proj with residual accumulate into x_cur
        gemm(y16, wout16, x_cur, ROWS, DM, ED, nullptr, 2);
    }

    // final FC + sigmoid -> out[4096]
    fc_sigmoid<<<ROWS, 256, 0, stream>>>(x_cur, fc_w, fc_b, out);
}
